// GraphConv2d_30124900614637
// MI455X (gfx1250) — compile-verified, matrix operand order checked by reading, not run on hardware
//
#include <hip/hip_runtime.h>

// ---------------------------------------------------------------------------
// EdgeConv (gather -> 3x3 conv -> BN(train) -> ReLU -> max over K) for MI455X
// Conv = implicit GEMM via v_wmma_f32_16x16x32_bf16.
// h tile staged in LDS with global_load_async_to_lds_b128 (ASYNCcnt path),
// shared by 4 waves; each wave computes 64 co x 64 spatial (16 accumulators).
// Conv output layout [b][n][k][co] -> v8f epilogue stores + coalesced tails.
// ---------------------------------------------------------------------------

typedef __bf16 bf16;
typedef __attribute__((ext_vector_type(16))) __bf16 v16bf;
typedef __attribute__((ext_vector_type(8)))  float  v8f;

constexpr int Bn    = 4;
constexpr int C_IN  = 128;
constexpr int Nn    = 4096;
constexpr int Kk    = 16;
constexpr int C_OUT = 256;
constexpr int CI2   = 2 * C_IN;      // 256 conv input channels
constexpr int NP    = Nn + 2;        // padded height
constexpr int KP    = Kk + 2;        // padded width
constexpr int RED   = CI2 * 9;       // 2304 reduction length (ci x 3 x 3)

// LDS tile: 6 rows x 18 cols x 256 ci bf16; pad (row,col) stride 512B -> 528B
constexpr int SEG_BYTES   = 528;                 // 264 bf16 per (row,col) segment
constexpr int N_SEG       = 6 * 18;              // 108 segments
constexpr int SMEM_BYTES  = N_SEG * SEG_BYTES;   // 57,024 B
constexpr int ASYNC_TOTAL = N_SEG * 32;          // 3456 b128 transfers (512B/seg)
constexpr int ASYNC_ITERS = ASYNC_TOTAL / 128;   // 27 per thread (128-thread block)

__device__ __forceinline__ bf16 f2bf(float f) {
    unsigned u = __builtin_bit_cast(unsigned, f);
    unsigned r = u + 0x7FFFu + ((u >> 16) & 1u);   // round-to-nearest-even
    unsigned short s = (unsigned short)(r >> 16);
    return __builtin_bit_cast(bf16, s);
}

// ---- Stage 1: gather x_i / (x_j - x_i) into padded bf16 h[b][n+1][k+1][ci] ----
__global__ void gather_build(const float* __restrict__ x, const float* __restrict__ y,
                             const int* __restrict__ ei, bf16* __restrict__ h) {
    int p = blockIdx.x;                 // 0 .. B*N*K-1
    int c = threadIdx.x;                // 0 .. 127
    int k = p % Kk;
    int n = (p / Kk) % Nn;
    int b = p / (Kk * Nn);
    int i0 = ei[(((size_t)0 * Bn + b) * Nn + n) * Kk + k];   // neighbor (for y)
    int i1 = ei[(((size_t)1 * Bn + b) * Nn + n) * Kk + k];   // center   (for x)
    float xi = x[((size_t)b * C_IN + c) * Nn + i1];
    float xj = y[((size_t)b * C_IN + c) * Nn + i0];
    size_t base = (((size_t)b * NP + (n + 1)) * KP + (k + 1)) * CI2;
    h[base + c]        = f2bf(xi);
    h[base + C_IN + c] = f2bf(xj - xi);
}

// ---- Stage 2: repack weights into WMMA-A layout: wA[co][tap*256 + ci] ----
__global__ void weight_prep(const float* __restrict__ w, bf16* __restrict__ wA) {
    int idx = blockIdx.x * blockDim.x + threadIdx.x;
    if (idx >= C_OUT * RED) return;
    int r   = idx % RED;
    int co  = idx / RED;
    int ci  = r % CI2;
    int tap = r / CI2;
    int dn = tap / 3, dk = tap % 3;
    wA[idx] = f2bf(w[(((size_t)co * CI2 + ci) * 3 + dn) * 3 + dk]);
}

// ---- Stage 3: implicit-GEMM 3x3 conv via WMMA bf16 + async LDS staging ----
// Block = 128 threads (4 wave32). blockIdx.x selects (b, n0=4*n-tile).
// Block output: 256 co x 64 spatial. Wave wv: co [wv*64, wv*64+64) x 64 spatial.
__global__ __launch_bounds__(128) void conv_wmma(const bf16* __restrict__ h,
                                                 const bf16* __restrict__ wA,
                                                 float* __restrict__ convo /* [b][n][k][co] f32 */) {
    __shared__ __align__(16) unsigned char smem[SMEM_BYTES];

    int tile = blockIdx.x;              // 0 .. B*N/4-1
    int n0 = (tile % (Nn / 4)) * 4;
    int b  = tile / (Nn / 4);
    int lane = threadIdx.x & 31;
    int wv   = threadIdx.x >> 5;        // 0..3
    int co_base = wv * 64;
    int lk = lane & 15;                 // spatial column / A-row select
    int hi = lane >> 4;                 // 0: K=0..15 half, 1: K=16..31 half

    // -------- async stage h rows n0..n0+5 (padded coords), cols 0..17 --------
    {
        unsigned smem_base = (unsigned)(unsigned long long)(uintptr_t)&smem[0];
        const bf16* gbase = h + ((size_t)(b * NP + n0) * KP) * CI2;
        int t = threadIdx.x;
        #pragma unroll 1
        for (int i = 0; i < ASYNC_ITERS; ++i) {
            int linear = i * 128 + t;                // 0 .. 3455
            int seg  = linear >> 5;                  // row*18 + col
            int part = linear & 31;                  // 16-byte chunk within 512B
            const bf16* gsrc = gbase + (size_t)seg * CI2 + part * 8;
            unsigned ldst = smem_base + seg * SEG_BYTES + part * 16;
            unsigned long long gaddr = (unsigned long long)(uintptr_t)gsrc;
            asm volatile("global_load_async_to_lds_b128 %0, %1, off"
                         :: "v"(ldst), "v"(gaddr) : "memory");
        }
        asm volatile("s_wait_asynccnt 0" ::: "memory");
    }
    __syncthreads();

    // -------- main MAC loops --------
    v8f acc[4][4];                       // [co-tile][spatial-tile]
    #pragma unroll
    for (int t = 0; t < 4; ++t)
        #pragma unroll
        for (int j = 0; j < 4; ++j)
            acc[t][j] = (v8f){};

    #pragma unroll 1
    for (int tap = 0; tap < 9; ++tap) {
        const int dn = tap / 3, dk = tap % 3;
        const bf16* wrow = wA + (size_t)tap * CI2 + hi * 16;
        const unsigned char* scol = smem + (size_t)(lk + dk) * SEG_BYTES + hi * 32;
        #pragma unroll
        for (int ch = 0; ch < 8; ++ch) {
            v16bf a0 = *(const v16bf*)(wrow + (size_t)(co_base +  0 + lk) * RED + ch * 32);
            v16bf a1 = *(const v16bf*)(wrow + (size_t)(co_base + 16 + lk) * RED + ch * 32);
            v16bf a2 = *(const v16bf*)(wrow + (size_t)(co_base + 32 + lk) * RED + ch * 32);
            v16bf a3 = *(const v16bf*)(wrow + (size_t)(co_base + 48 + lk) * RED + ch * 32);
            v16bf b0 = *(const v16bf*)(scol + (size_t)(0 + dn) * 18 * SEG_BYTES + ch * 64);
            v16bf b1 = *(const v16bf*)(scol + (size_t)(1 + dn) * 18 * SEG_BYTES + ch * 64);
            v16bf b2 = *(const v16bf*)(scol + (size_t)(2 + dn) * 18 * SEG_BYTES + ch * 64);
            v16bf b3 = *(const v16bf*)(scol + (size_t)(3 + dn) * 18 * SEG_BYTES + ch * 64);
            acc[0][0] = __builtin_amdgcn_wmma_f32_16x16x32_bf16(false, a0, false, b0, (short)0, acc[0][0], false, false);
            acc[1][0] = __builtin_amdgcn_wmma_f32_16x16x32_bf16(false, a1, false, b0, (short)0, acc[1][0], false, false);
            acc[2][0] = __builtin_amdgcn_wmma_f32_16x16x32_bf16(false, a2, false, b0, (short)0, acc[2][0], false, false);
            acc[3][0] = __builtin_amdgcn_wmma_f32_16x16x32_bf16(false, a3, false, b0, (short)0, acc[3][0], false, false);
            acc[0][1] = __builtin_amdgcn_wmma_f32_16x16x32_bf16(false, a0, false, b1, (short)0, acc[0][1], false, false);
            acc[1][1] = __builtin_amdgcn_wmma_f32_16x16x32_bf16(false, a1, false, b1, (short)0, acc[1][1], false, false);
            acc[2][1] = __builtin_amdgcn_wmma_f32_16x16x32_bf16(false, a2, false, b1, (short)0, acc[2][1], false, false);
            acc[3][1] = __builtin_amdgcn_wmma_f32_16x16x32_bf16(false, a3, false, b1, (short)0, acc[3][1], false, false);
            acc[0][2] = __builtin_amdgcn_wmma_f32_16x16x32_bf16(false, a0, false, b2, (short)0, acc[0][2], false, false);
            acc[1][2] = __builtin_amdgcn_wmma_f32_16x16x32_bf16(false, a1, false, b2, (short)0, acc[1][2], false, false);
            acc[2][2] = __builtin_amdgcn_wmma_f32_16x16x32_bf16(false, a2, false, b2, (short)0, acc[2][2], false, false);
            acc[3][2] = __builtin_amdgcn_wmma_f32_16x16x32_bf16(false, a3, false, b2, (short)0, acc[3][2], false, false);
            acc[0][3] = __builtin_amdgcn_wmma_f32_16x16x32_bf16(false, a0, false, b3, (short)0, acc[0][3], false, false);
            acc[1][3] = __builtin_amdgcn_wmma_f32_16x16x32_bf16(false, a1, false, b3, (short)0, acc[1][3], false, false);
            acc[2][3] = __builtin_amdgcn_wmma_f32_16x16x32_bf16(false, a2, false, b3, (short)0, acc[2][3], false, false);
            acc[3][3] = __builtin_amdgcn_wmma_f32_16x16x32_bf16(false, a3, false, b3, (short)0, acc[3][3], false, false);
        }
    }

    // Epilogue: lane holds 8 consecutive co (r=0..7 at co_base + t*16 + hi*8).
    // convo layout [b][n][k][co] -> one 32B v8f store per accumulator.
    #pragma unroll
    for (int t = 0; t < 4; ++t) {
        #pragma unroll
        for (int j = 0; j < 4; ++j) {
            float* ob = convo + ((size_t)(b * Nn + n0 + j) * Kk + lk) * C_OUT
                              + co_base + t * 16 + hi * 8;
            *(v8f*)ob = acc[t][j];
        }
    }
}

// ---- Stage 4: per-channel sum / sum-of-squares (bias dropped: BN cancels it) ----
__global__ void stats(const float* __restrict__ convo, float* __restrict__ sums) {
    int co = threadIdx.x;              // 256 threads, one channel each
    int slab = blockIdx.x;             // 256 slabs of 64 (b,n) rows
    float s = 0.f, s2 = 0.f;
    for (int rr = 0; rr < 64; ++rr) {
        size_t row = (size_t)slab * 64 + rr;
        const float* p = convo + row * Kk * C_OUT + co;   // [row][k][co]
        #pragma unroll
        for (int k = 0; k < Kk; ++k) { float v = p[(size_t)k * C_OUT]; s += v; s2 += v * v; }
    }
    atomicAdd(&sums[co], s);
    atomicAdd(&sums[C_OUT + co], s2);
}

// ---- Stage 5: BN(train) + ReLU + max over K -> out[b][co][n] ----
__global__ void finalize(const float* __restrict__ convo, const float* __restrict__ sums,
                         const float* __restrict__ gamma, const float* __restrict__ beta,
                         float* __restrict__ out) {
    int co = threadIdx.x;              // 256
    size_t row = blockIdx.x;           // b*N + n
    int n = (int)(row % Nn);
    int b = (int)(row / Nn);
    const float Minv = 1.0f / (float)(Bn * Nn * Kk);
    float mean = sums[co] * Minv;
    float var  = sums[C_OUT + co] * Minv - mean * mean;
    float sc = gamma[co] * rsqrtf(var + 1e-5f);
    float sh = beta[co] - mean * sc;
    const float* p = convo + row * Kk * C_OUT + co;       // [row][k][co]
    float m = 0.f;                     // ReLU output >= 0, so 0 is a valid identity
    #pragma unroll
    for (int k = 0; k < Kk; ++k) {
        float v = sc * p[(size_t)k * C_OUT] + sh;
        v = v > 0.f ? v : 0.f;
        m = v > m ? v : m;
    }
    out[((size_t)b * C_OUT + co) * Nn + n] = m;
}

extern "C" void kernel_launch(void* const* d_in, const int* in_sizes, int n_in,
                              void* d_out, int out_size, void* d_ws, size_t ws_size,
                              hipStream_t stream) {
    const float* x     = (const float*)d_in[0];
    const float* y     = (const float*)d_in[1];
    const int*   ei    = (const int*)d_in[2];
    const float* cw    = (const float*)d_in[3];
    // d_in[4] = conv bias: cancels exactly under training-mode BatchNorm
    const float* gamma = (const float*)d_in[5];
    const float* beta  = (const float*)d_in[6];
    float* out = (float*)d_out;

    char* ws = (char*)d_ws;
    const size_t hbytes = (size_t)Bn * NP * KP * CI2 * sizeof(bf16);      // ~151 MB
    const size_t cbytes = (size_t)Bn * Nn * C_OUT * Kk * sizeof(float);   // 256 MB
    const size_t wbytes = (size_t)C_OUT * RED * sizeof(bf16);             // ~1.2 MB
    bf16*  h     = (bf16*)ws;
    float* convo = (float*)(ws + hbytes);
    bf16*  wA    = (bf16*)(ws + hbytes + cbytes);
    float* sums  = (float*)(ws + hbytes + cbytes + wbytes);

    hipMemsetAsync(h, 0, hbytes, stream);                    // zero padded halo
    hipMemsetAsync(sums, 0, 2 * C_OUT * sizeof(float), stream);

    gather_build<<<Bn * Nn * Kk, C_IN, 0, stream>>>(x, y, ei, h);
    weight_prep<<<(C_OUT * RED + 255) / 256, 256, 0, stream>>>(cw, wA);
    conv_wmma<<<Bn * Nn / 4, 128, 0, stream>>>(h, wA, convo);
    stats<<<256, 256, 0, stream>>>(convo, sums);
    finalize<<<Bn * Nn, 256, 0, stream>>>(convo, sums, gamma, beta, out);
}
